// Attention_62672162783397
// MI455X (gfx1250) — matrix operand order repaired and checked by reading, not compiled, not run on hardware
//
#include <hip/hip_runtime.h>

#define BB 4
#define SS 2048
#define DD 1024
#define NH 16
#define NKV 4
#define HD 64
#define MM (BB*SS)   // 8192 rows

typedef __bf16 bf16;
typedef __attribute__((ext_vector_type(16))) __bf16 v16bf;
typedef __attribute__((ext_vector_type(8)))  __bf16 bf16x8;
typedef __attribute__((ext_vector_type(8)))  float  v8f;

union AF { v16bf v; bf16x8 h[2]; };

// Async copy of 16 bytes global -> LDS (CDNA5 GLOBAL_LOAD_ASYNC_TO_LDS_B128,
// GVS mode: saddr base + 32-bit vgpr byte offset). Tracked by ASYNCcnt.
__device__ __forceinline__ void async_b128(const void* gbase, int gbyteoff, void* ldsp) {
    unsigned lds = (unsigned)(uintptr_t)ldsp;     // flat-shared addr: low 32 = LDS offset
    asm volatile("global_load_async_to_lds_b128 %0, %1, %2"
                 :: "v"(lds), "v"(gbyteoff),
                    "s"((unsigned long long)(uintptr_t)gbase)
                 : "memory");
}
__device__ __forceinline__ void wait_async0() {
    asm volatile("s_wait_asynccnt 0x0" ::: "memory");
}

// ---- 16-lane (half-wave) reductions via DPP16: pure VALU, no LDS waits ----
template<int CTRL>
__device__ __forceinline__ float dpp_f(float x) {
    return __int_as_float(
        __builtin_amdgcn_update_dpp(0, __float_as_int(x), CTRL, 0xF, 0xF, true));
}
__device__ __forceinline__ float red16_max(float x) {
    x = fmaxf(x, dpp_f<0xB1>(x));    // quad_perm [1,0,3,2]  (xor 1)
    x = fmaxf(x, dpp_f<0x4E>(x));    // quad_perm [2,3,0,1]  (xor 2)
    x = fmaxf(x, dpp_f<0x141>(x));   // row_half_mirror      (xor 7)
    x = fmaxf(x, dpp_f<0x140>(x));   // row_mirror           (xor 15)
    return x;
}
__device__ __forceinline__ float red16_sum(float x) {
    x += dpp_f<0xB1>(x);
    x += dpp_f<0x4E>(x);
    x += dpp_f<0x141>(x);
    x += dpp_f<0x140>(x);
    return x;
}

// ---------------------------------------------------------------- convert ---
__global__ void cvt_f32_bf16(const float* __restrict__ in, bf16* __restrict__ out, int n) {
    int i = blockIdx.x * blockDim.x + threadIdx.x;
    int stride = gridDim.x * blockDim.x;
    for (; i < n; i += stride) out[i] = (bf16)in[i];
}

// in: K x N row-major fp32 -> out: N x K row-major bf16 (weight pre-transpose)
__global__ void cvt_transpose_bf16(const float* __restrict__ in, bf16* __restrict__ out,
                                   int K, int N) {
    int i = blockIdx.x * blockDim.x + threadIdx.x;
    int total = K * N;
    int stride = gridDim.x * blockDim.x;
    for (; i < total; i += stride) {
        int n = i / K, k = i - n * K;
        out[i] = (bf16)in[(size_t)k * N + n];
    }
}

// ------------------------------------------------------------------- GEMM ---
// C[M,N] = A[M,K] @ Bt[N,K]^T (+bias)(+RoPE), bf16 in, fp32 accum.
// EPI 0: fp32 row-major out.
// EPI 1: bias+RoPE, *oscale, bf16 out (b,head,s,hd).
// EPI 2: bias only,  bf16 out transposed (b,head,hd,s)  [for V].
template<int EPI>
__global__ __launch_bounds__(256)
void gemm_wmma(const bf16* __restrict__ A, const bf16* __restrict__ Bt,
               const float* __restrict__ bias, const float* __restrict__ rope,
               void* __restrict__ outp, int N, int K, float oscale)
{
    __shared__ __align__(16) bf16 As[2][128][40];   // [buf][row][k]
    __shared__ __align__(16) bf16 Bs[2][128][40];   // [buf][n][k] (Bt rows are n)

    const int tid   = threadIdx.x;
    const int w     = tid >> 5, lane = tid & 31;
    const int lh    = lane >> 4, laneN = lane & 15;
    const int wm    = w & 3, wn = w >> 2;          // 4 waves M x 2 waves N
    const int mBase = blockIdx.y * 128;
    const int nBase = blockIdx.x * 128;

    const v8f vzero = {0.f,0.f,0.f,0.f,0.f,0.f,0.f,0.f};
    v8f acc[2][4];
    #pragma unroll
    for (int mt = 0; mt < 2; ++mt)
        #pragma unroll
        for (int nt = 0; nt < 4; ++nt) acc[mt][nt] = vzero;

    const int kA0 = lh ? 8 : 0;    // A-frag K chunks {0-7,16-23}/{8-15,24-31}
    const int kB0 = lh ? 16 : 0;   // unused; B-frag reads the same K chunks as A (kA0)
    const int row_f = tid >> 2, kc_f = (tid & 3) << 3;   // fill coords (2 chunks/thread)

    auto issue_tiles = [&](int buf, int kb) {
        #pragma unroll
        for (int r = 0; r < 2; ++r) {
            int row = row_f + r * 64;
            async_b128(A,  (int)(((size_t)(mBase + row) * K + kb + kc_f) * 2), &As[buf][row][kc_f]);
            async_b128(Bt, (int)(((size_t)(nBase + row) * K + kb + kc_f) * 2), &Bs[buf][row][kc_f]);
        }
    };

    issue_tiles(0, 0);
    wait_async0();
    __syncthreads();

    const int nsteps = K >> 5;
    for (int step = 0; step < nsteps; ++step) {
        const int buf = step & 1;
        if (step + 1 < nsteps) issue_tiles(buf ^ 1, (step + 1) << 5);

        AF af[2];
        #pragma unroll
        for (int mt = 0; mt < 2; ++mt) {
            int row = wm * 32 + mt * 16 + laneN;
            af[mt].h[0] = *(const bf16x8*)&As[buf][row][kA0];
            af[mt].h[1] = *(const bf16x8*)&As[buf][row][kA0 + 16];
        }
        #pragma unroll
        for (int nt = 0; nt < 4; ++nt) {
            AF bfr;
            int col = wn * 64 + nt * 16 + laneN;
            bfr.h[0] = *(const bf16x8*)&Bs[buf][col][kA0];
            bfr.h[1] = *(const bf16x8*)&Bs[buf][col][kA0 + 16];
            #pragma unroll
            for (int mt = 0; mt < 2; ++mt)
                acc[mt][nt] = __builtin_amdgcn_wmma_f32_16x16x32_bf16(
                    false, af[mt].v, false, bfr.v, (short)0, acc[mt][nt], false, false);
        }
        wait_async0();
        __syncthreads();
    }

    // ----------------------------- epilogue -----------------------------
    if constexpr (EPI == 0) {
        float* out = (float*)outp;
        #pragma unroll
        for (int mt = 0; mt < 2; ++mt)
            #pragma unroll
            for (int nt = 0; nt < 4; ++nt)
                #pragma unroll
                for (int v = 0; v < 8; ++v) {
                    int gm = mBase + wm * 32 + mt * 16 + v + lh * 8;
                    int gn = nBase + wn * 64 + nt * 16 + laneN;
                    out[(size_t)gm * N + gn] = acc[mt][nt][v];
                }
    } else {
        float biasv[4];
        #pragma unroll
        for (int nt = 0; nt < 4; ++nt)
            biasv[nt] = bias[nBase + wn * 64 + nt * 16 + laneN];
        bf16* out = (bf16*)outp;
        const int HN   = N / HD;
        const int head = (nBase + wn * 64) / HD;   // wave N-tile == one head
        if constexpr (EPI == 1) {
            #pragma unroll
            for (int mt = 0; mt < 2; ++mt)
                #pragma unroll
                for (int v = 0; v < 8; ++v) {
                    int gm = mBase + wm * 32 + mt * 16 + v + lh * 8;
                    int bidx = gm / SS, spos = gm % SS;
                    size_t dst = (((size_t)bidx * HN + head) * SS + spos) * HD;
                    #pragma unroll
                    for (int ntp = 0; ntp < 2; ++ntp) {    // (nt, nt+2) = (hd, hd+32)
                        int hd1 = ntp * 16 + laneN;        // < 32
                        float f = rope[(size_t)gm * (HD / 2) + hd1];
                        float sn, cs; __sincosf(f, &sn, &cs);
                        float x1 = acc[mt][ntp][v]     + biasv[ntp];
                        float x2 = acc[mt][ntp + 2][v] + biasv[ntp + 2];
                        out[dst + hd1]      = (bf16)((x1 * cs - x2 * sn) * oscale);
                        out[dst + hd1 + 32] = (bf16)((x2 * cs + x1 * sn) * oscale);
                    }
                }
        } else {  // EPI == 2 : V, stored transposed (b, head, hd, s)
            #pragma unroll
            for (int mt = 0; mt < 2; ++mt)
                #pragma unroll
                for (int nt = 0; nt < 4; ++nt)
                    #pragma unroll
                    for (int v = 0; v < 8; ++v) {
                        int gm = mBase + wm * 32 + mt * 16 + v + lh * 8;
                        int bidx = gm / SS, spos = gm % SS;
                        int hd = nt * 16 + laneN;
                        out[(((size_t)bidx * HN + head) * HD + hd) * SS + spos] =
                            (bf16)(acc[mt][nt][v] + biasv[nt]);
                    }
        }
    }
}

// -------------------------------------------------------- flash attention ---
// grid = (B*NH, S/128); block 256 = 8 waves, each wave owns 16 query rows.
// Q pre-scaled by HD^-0.5 in projection.  V arrives transposed (b,kvh,hd,s).
__global__ __launch_bounds__(256)
void attn_wmma(const bf16* __restrict__ Q, const bf16* __restrict__ Km,
               const bf16* __restrict__ Vt, bf16* __restrict__ Om)
{
    __shared__ __align__(16) bf16 Ks[2][64][72];   // [buf][kv][hd]
    __shared__ __align__(16) bf16 Vs[2][64][72];   // [buf][hd][kv]
    __shared__ __align__(16) bf16 Ps[8][16][72];   // per-wave P C->A relayout

    const int tid  = threadIdx.x;
    const int w    = tid >> 5, lane = tid & 31;
    const int lh   = lane >> 4, laneN = lane & 15;
    const int bh   = blockIdx.x;
    const int b    = bh / NH, h = bh % NH;
    const int kvh  = h / (NH / NKV);
    const int qbase = blockIdx.y * 128 + w * 16;

    const bf16* Qp  = Q  + ((size_t)(b * NH  + h)   * SS) * HD;
    const bf16* Kp  = Km + ((size_t)(b * NKV + kvh) * SS) * HD;
    const bf16* Vtp = Vt + ((size_t)(b * NKV + kvh) * HD) * SS;

    const int kA0 = lh ? 8 : 0;
    const int kB0 = lh ? 16 : 0;
    const int row_f = tid >> 3, x0_f = (tid & 7) << 3;   // fill coords

    auto issue_kv = [&](int buf, int kvb) {
        #pragma unroll
        for (int r = 0; r < 2; ++r) {
            int row = row_f + r * 32;
            async_b128(Kp,  (int)(((size_t)(kvb + row) * HD + x0_f) * 2), &Ks[buf][row][x0_f]);
            async_b128(Vtp, (int)(((size_t)row * SS + kvb + x0_f) * 2),   &Vs[buf][row][x0_f]);
        }
    };

    AF qf[2];
    {
        int row = qbase + laneN;
        #pragma unroll
        for (int kk = 0; kk < 2; ++kk) {
            qf[kk].h[0] = *(const bf16x8*)&Qp[(size_t)row * HD + kk * 32 + kA0];
            qf[kk].h[1] = *(const bf16x8*)&Qp[(size_t)row * HD + kk * 32 + kA0 + 16];
        }
    }

    const v8f vzero = {0.f,0.f,0.f,0.f,0.f,0.f,0.f,0.f};
    v8f o[4];
    #pragma unroll
    for (int nt = 0; nt < 4; ++nt) o[nt] = vzero;
    float mrow[8], lrow[8];
    #pragma unroll
    for (int v = 0; v < 8; ++v) { mrow[v] = -1e30f; lrow[v] = 0.f; }

    issue_kv(0, 0);
    wait_async0();
    __syncthreads();

    for (int kvb = 0; kvb < SS; kvb += 64) {
        const int buf = (kvb >> 6) & 1;
        if (kvb + 64 < SS) issue_kv(buf ^ 1, kvb + 64);

        // scores = Q @ K^T  (16 x 64)
        v8f sacc[4];
        #pragma unroll
        for (int nt = 0; nt < 4; ++nt) sacc[nt] = vzero;
        #pragma unroll
        for (int kk = 0; kk < 2; ++kk)
            #pragma unroll
            for (int nt = 0; nt < 4; ++nt) {
                AF bfr;
                int n = nt * 16 + laneN;
                bfr.h[0] = *(const bf16x8*)&Ks[buf][n][kk * 32 + kA0];
                bfr.h[1] = *(const bf16x8*)&Ks[buf][n][kk * 32 + kA0 + 16];
                sacc[nt] = __builtin_amdgcn_wmma_f32_16x16x32_bf16(
                    false, qf[kk].v, false, bfr.v, (short)0, sacc[nt], false, false);
            }

        // ---- online softmax: DPP16 butterflies (VALU only, no LDS) ----
        float corr[8];
        #pragma unroll
        for (int v = 0; v < 8; ++v) {
            float tv = fmaxf(fmaxf(sacc[0][v], sacc[1][v]),
                             fmaxf(sacc[2][v], sacc[3][v]));
            tv = red16_max(tv);
            float newm = fmaxf(mrow[v], tv);
            corr[v] = __expf(mrow[v] - newm);
            mrow[v] = newm;
        }
        #pragma unroll
        for (int nt = 0; nt < 4; ++nt)
            #pragma unroll
            for (int v = 0; v < 8; ++v)
                sacc[nt][v] = __expf(sacc[nt][v] - mrow[v]);
        #pragma unroll
        for (int v = 0; v < 8; ++v) {
            float rs = (sacc[0][v] + sacc[1][v]) + (sacc[2][v] + sacc[3][v]);
            rs = red16_sum(rs);
            lrow[v] = lrow[v] * corr[v] + rs;
        }
        #pragma unroll
        for (int nt = 0; nt < 4; ++nt)
            #pragma unroll
            for (int v = 0; v < 8; ++v) o[nt][v] *= corr[v];

        // C-layout -> A-fragment layout via per-wave LDS buffer
        #pragma unroll
        for (int nt = 0; nt < 4; ++nt)
            #pragma unroll
            for (int v = 0; v < 8; ++v)
                Ps[w][v + lh * 8][nt * 16 + laneN] = (bf16)sacc[nt][v];

        // O += P @ V
        #pragma unroll
        for (int kk = 0; kk < 2; ++kk) {
            AF pf;
            pf.h[0] = *(const bf16x8*)&Ps[w][laneN][kk * 32 + kA0];
            pf.h[1] = *(const bf16x8*)&Ps[w][laneN][kk * 32 + kA0 + 16];
            #pragma unroll
            for (int nt = 0; nt < 4; ++nt) {
                AF vfr;
                int n = nt * 16 + laneN;
                vfr.h[0] = *(const bf16x8*)&Vs[buf][n][kk * 32 + kA0];
                vfr.h[1] = *(const bf16x8*)&Vs[buf][n][kk * 32 + kA0 + 16];
                o[nt] = __builtin_amdgcn_wmma_f32_16x16x32_bf16(
                    false, pf.v, false, vfr.v, (short)0, o[nt], false, false);
            }
        }
        wait_async0();
        __syncthreads();
    }

    // normalize + write (b, s, h*64+hd) bf16
    #pragma unroll
    for (int v = 0; v < 8; ++v) {
        float inv = 1.f / lrow[v];
        int spos = qbase + v + lh * 8;
        size_t dst = ((size_t)b * SS + spos) * DD + h * HD;
        #pragma unroll
        for (int nt = 0; nt < 4; ++nt)
            Om[dst + nt * 16 + laneN] = (bf16)(o[nt][v] * inv);
    }
}

// ----------------------------------------------------------------- launch ---
extern "C" void kernel_launch(void* const* d_in, const int* in_sizes, int n_in,
                              void* d_out, int out_size, void* d_ws, size_t ws_size,
                              hipStream_t stream)
{
    const float* hidden = (const float*)d_in[0];
    const float* rope   = (const float*)d_in[1];
    const float* Wq = (const float*)d_in[2];
    const float* bq = (const float*)d_in[3];
    const float* Wk = (const float*)d_in[4];
    const float* bk = (const float*)d_in[5];
    const float* Wv = (const float*)d_in[6];
    const float* bv = (const float*)d_in[7];
    const float* Wo = (const float*)d_in[8];
    float* out = (float*)d_out;

    char* p = (char*)d_ws;
    bf16* hid_b = (bf16*)p; p += (size_t)MM * DD * 2;
    bf16* WqT   = (bf16*)p; p += (size_t)DD * DD * 2;            // 1024 x 1024 (NxK)
    bf16* WkT   = (bf16*)p; p += (size_t)DD * (NKV * HD) * 2;    // 256  x 1024
    bf16* WvT   = (bf16*)p; p += (size_t)DD * (NKV * HD) * 2;
    bf16* WoT   = (bf16*)p; p += (size_t)DD * DD * 2;
    bf16* Qb    = (bf16*)p; p += (size_t)MM * DD * 2;            // (b,h,s,hd)
    bf16* Kb    = (bf16*)p; p += (size_t)MM * (NKV * HD) * 2;    // (b,kvh,s,hd)
    bf16* Vb    = (bf16*)p; p += (size_t)MM * (NKV * HD) * 2;    // (b,kvh,hd,s)
    bf16* Ab    = (bf16*)p; p += (size_t)MM * DD * 2;            // (b,s,1024)

    const float qscale = 0.125f;   // HD^-0.5 folded into Q

    cvt_f32_bf16<<<1024, 256, 0, stream>>>(hidden, hid_b, MM * DD);
    cvt_transpose_bf16<<<512, 256, 0, stream>>>(Wq, WqT, DD, DD);
    cvt_transpose_bf16<<<256, 256, 0, stream>>>(Wk, WkT, DD, NKV * HD);
    cvt_transpose_bf16<<<256, 256, 0, stream>>>(Wv, WvT, DD, NKV * HD);
    cvt_transpose_bf16<<<512, 256, 0, stream>>>(Wo, WoT, DD, DD);

    gemm_wmma<1><<<dim3(DD / 128, MM / 128), 256, 0, stream>>>(
        hid_b, WqT, bq, rope, Qb, DD, DD, qscale);
    gemm_wmma<1><<<dim3((NKV * HD) / 128, MM / 128), 256, 0, stream>>>(
        hid_b, WkT, bk, rope, Kb, NKV * HD, DD, 1.0f);
    gemm_wmma<2><<<dim3((NKV * HD) / 128, MM / 128), 256, 0, stream>>>(
        hid_b, WvT, bv, nullptr, Vb, NKV * HD, DD, 1.0f);

    attn_wmma<<<dim3(BB * NH, SS / 128), 256, 0, stream>>>(Qb, Kb, Vb, Ab);

    gemm_wmma<0><<<dim3(DD / 128, MM / 128), 256, 0, stream>>>(
        Ab, WoT, nullptr, nullptr, out, DD, DD, 1.0f);
}
